// MultiHeadAttention_18554258719484
// MI455X (gfx1250) — compile-verified, matrix operand order checked by reading, not run on hardware
//
#include <hip/hip_runtime.h>

// ---------------------------------------------------------------------------
// MHA forward for MI455X (gfx1250, wave32, WMMA bf16 path)
// S=2048, B=2, D_MODEL=1024, H=16, DH=64
//
// ~100 GFLOP vs ~130 MB HBM traffic -> compute-bound: everything on
// V_WMMA_F32_16X16X32_BF16 (bf16 operands, f32 accum).  Weights are
// pre-converted to bf16 and transposed once so every WMMA operand fetch is a
// contiguous 16-byte load matching the CDNA5 16-bit A/B VGPR layout.
// GEMM k-loops are software-pipelined with ping-pong buffers (unroll-by-2,
// no register-rotation copies).
// ---------------------------------------------------------------------------

typedef __attribute__((ext_vector_type(16))) __bf16 v16bf;
typedef __attribute__((ext_vector_type(8)))  __bf16 v8bf;
typedef __attribute__((ext_vector_type(8)))  float  v8f;

#define S_LEN 2048
#define BATCH 2
#define DM    1024
#define NH    16
#define DHD   64
#define MROWS (S_LEN * BATCH)   // 4096
#define QK_SCALE 0.125f          // 1/sqrt(64)
#define NEG_BIG (-1.0e30f)

// fp32 -> bf16 (hardware v_cvt_pk_bf16_f32, RNE)
__device__ __forceinline__ __bf16 f2bf(float f) { return (__bf16)f; }

// 16-bit A/B operand layout per lane (CDNA5 ISA 7.12.2):
// halves 0..7  -> K = kb + 0..7
// halves 8..15 -> K = kb + 16 + 0..7        (kb = 8 * (lane>>4), folded into p)
__device__ __forceinline__ v16bf load_tile_bf16(const __bf16* p) {
    v8bf lo = *(const v8bf*)(p);
    v8bf hi = *(const v8bf*)(p + 16);
    return __builtin_shufflevector(lo, hi, 0,1,2,3,4,5,6,7,8,9,10,11,12,13,14,15);
}

__device__ __forceinline__ v16bf load_tile_f32(const float* p) {
    v16bf r;
#pragma unroll
    for (int i = 0; i < 8; ++i) { r[i] = f2bf(p[i]); r[i + 8] = f2bf(p[16 + i]); }
    return r;
}

__device__ __forceinline__ v8f wmma_bf16(v16bf a, v16bf b, v8f c) {
    return __builtin_amdgcn_wmma_f32_16x16x32_bf16(
        /*neg_a=*/false, a, /*neg_b=*/false, b,
        /*c_mod=*/(short)0, c, /*reuse_a=*/false, /*reuse_b=*/false);
}

// ---------------------------------------------------------------------------
// Stage 0: weight convert + transpose.  W[k][n] f32 -> Wt[n][k] bf16.
// LDS-tiled 32x32 transpose, 256 threads (32x8).
// ---------------------------------------------------------------------------
__global__ __launch_bounds__(256)
void wcvt_kernel(const float* __restrict__ W, __bf16* __restrict__ Wt)
{
    __shared__ float tile[32][33];
    const int k0 = blockIdx.x * 32;
    const int n0 = blockIdx.y * 32;
    const int tx = threadIdx.x & 31;
    const int ty = threadIdx.x >> 5;     // 0..7
#pragma unroll
    for (int i = 0; i < 32; i += 8)
        tile[ty + i][tx] = W[(size_t)(k0 + ty + i) * DM + n0 + tx];
    __syncthreads();
#pragma unroll
    for (int i = 0; i < 32; i += 8)
        Wt[(size_t)(n0 + ty + i) * DM + k0 + tx] = f2bf(tile[tx][ty + i]);
}

// ---------------------------------------------------------------------------
// Stage 1: QKV projection.  out = X @ W + bias, cast to bf16.
// One wave per block; wave computes a 16x64 strip (4 N-tiles).
// K-loop 1024/32, ping-pong pipelined: phase 0 consumes buf0 while buf1
// loads, phase 1 consumes buf1 while buf0 loads.  Last pair peeled.
// vt==0: out[(b*H+h)*S + s][d]   (Q, K layout)
// vt==1: out[(b*H+h)*DH + d][s]  (V transposed for contiguous PV B-operands)
// ---------------------------------------------------------------------------
__global__ __launch_bounds__(32)
void qkv_proj_kernel(const float* __restrict__ X, const __bf16* __restrict__ Wt,
                     const float* __restrict__ bias, __bf16* __restrict__ out, int vt)
{
    const int lane  = threadIdx.x;
    const int half  = lane >> 4;
    const int l15   = lane & 15;
    const int kb    = 8 * half;
    const int mBase = blockIdx.x * 16;   // 0..4095
    const int nBase = blockIdx.y * 64;   // 0..1023

    v8f acc[4] = {};
    const float* arow = X + (size_t)(mBase + l15) * DM + kb;
    const __bf16* brow[4];
#pragma unroll
    for (int t = 0; t < 4; ++t)
        brow[t] = Wt + (size_t)(nBase + t * 16 + l15) * DM + kb;

    // prologue: chunk 0 into buffer 0
    v16bf a0 = load_tile_f32(arow);
    v16bf b0[4];
#pragma unroll
    for (int t = 0; t < 4; ++t) b0[t] = load_tile_bf16(brow[t]);
    v16bf a1, b1[4];

    int kk = 0;
    for (; kk < DM - 64; kk += 64) {
        // phase 0: load chunk kk+32 -> buf1, consume buf0
        __builtin_prefetch(arow + kk + 96, 0, 3);            // global_prefetch_b8
        a1 = load_tile_f32(arow + kk + 32);
#pragma unroll
        for (int t = 0; t < 4; ++t) b1[t] = load_tile_bf16(brow[t] + kk + 32);
#pragma unroll
        for (int t = 0; t < 4; ++t) acc[t] = wmma_bf16(a0, b0[t], acc[t]);
        // phase 1: load chunk kk+64 -> buf0, consume buf1
        a0 = load_tile_f32(arow + kk + 64);
#pragma unroll
        for (int t = 0; t < 4; ++t) b0[t] = load_tile_bf16(brow[t] + kk + 64);
#pragma unroll
        for (int t = 0; t < 4; ++t) acc[t] = wmma_bf16(a1, b1[t], acc[t]);
    }
    // epilogue pair: kk == DM-64
    a1 = load_tile_f32(arow + kk + 32);
#pragma unroll
    for (int t = 0; t < 4; ++t) b1[t] = load_tile_bf16(brow[t] + kk + 32);
#pragma unroll
    for (int t = 0; t < 4; ++t) acc[t] = wmma_bf16(a0, b0[t], acc[t]);
#pragma unroll
    for (int t = 0; t < 4; ++t) acc[t] = wmma_bf16(a1, b1[t], acc[t]);

#pragma unroll
    for (int t = 0; t < 4; ++t) {
        const int n  = nBase + t * 16 + l15;     // = h*64 + d
        const float bvv = bias[n];
        const int h = n >> 6, d = n & 63;
#pragma unroll
        for (int r = 0; r < 8; ++r) {
            const int m  = mBase + r + 8 * half; // = s*B + b
            const int s  = m >> 1, bb = m & 1;
            const int bh = bb * NH + h;
            const float val = acc[t][r] + bvv;
            size_t idx = vt ? ((size_t)bh * DHD + d) * S_LEN + s
                            : ((size_t)bh * S_LEN + s) * DHD + d;
            out[idx] = f2bf(val);
        }
    }
}

// ---------------------------------------------------------------------------
// Stage 2: causal flash attention.  One wave = one 16-row query tile of one
// batch-head.  Online softmax over sk chunks of 32.  V-tile loads are issued
// before the softmax VALU block so exp/max math hides their latency.
// ---------------------------------------------------------------------------
__global__ __launch_bounds__(128)
void attn_kernel(const __bf16* __restrict__ Q, const __bf16* __restrict__ K,
                 const __bf16* __restrict__ Vt, __bf16* __restrict__ Xo)
{
    const int lane = threadIdx.x & 31;
    const int w    = threadIdx.x >> 5;
    const int gw   = blockIdx.x * 4 + w;     // 0..4095
    const int bh   = gw >> 7;                // 0..31  (= b*H + h)
    const int sqB  = (gw & 127) * 16;        // query tile base
    const int half = lane >> 4, l15 = lane & 15, kb = 8 * half;

    __shared__ __bf16 ldsP[4 * 16 * 32];     // per-wave 16x32 bf16 P tile
    __bf16* P = ldsP + w * (16 * 32);

    const __bf16* Qb = Q  + (size_t)bh * S_LEN * DHD;
    const __bf16* Kb = K  + (size_t)bh * S_LEN * DHD;
    const __bf16* Vb = Vt + (size_t)bh * DHD * S_LEN;

    // Q A-operand: 16 rows x 64 cols -> two K-steps, loaded once
    const __bf16* qrow = Qb + (size_t)(sqB + l15) * DHD;
    const v16bf qa0 = load_tile_bf16(qrow + kb);
    const v16bf qa1 = load_tile_bf16(qrow + 32 + kb);

    v8f o[4] = {};
    float mrow[8], lrow[8];
#pragma unroll
    for (int r = 0; r < 8; ++r) { mrow[r] = NEG_BIG; lrow[r] = 0.0f; }

    const int nChunks = (sqB + 15) / 32 + 1;
    for (int c = 0; c < nChunks; ++c) {
        const int skB = c * 32;

        // ---- scores: two 16x16 tiles, K-dim 64 = 2 WMMA each ----
        const __bf16* kr0 = Kb + (size_t)(skB + l15) * DHD;
        const __bf16* kr1 = Kb + (size_t)(skB + 16 + l15) * DHD;
        __builtin_prefetch(kr0 + 32 * DHD, 0, 3);            // next-chunk K rows
        v8f z = {};
        v8f s0 = wmma_bf16(qa0, load_tile_bf16(kr0 + kb), z);
        s0     = wmma_bf16(qa1, load_tile_bf16(kr0 + 32 + kb), s0);
        v8f s1 = wmma_bf16(qa0, load_tile_bf16(kr1 + kb), z);
        s1     = wmma_bf16(qa1, load_tile_bf16(kr1 + 32 + kb), s1);

        // ---- issue V-tile loads early: softmax math below hides latency ----
        v16bf vb[4];
#pragma unroll
        for (int t = 0; t < 4; ++t) {
            const __bf16* vr = Vb + (size_t)(t * 16 + l15) * S_LEN + skB + kb;
            vb[t] = load_tile_bf16(vr);
        }

        // ---- scale + causal mask + online softmax ----
        float p0[8], p1[8];
#pragma unroll
        for (int r = 0; r < 8; ++r) {
            const int sq = sqB + r + 8 * half;
            float x0 = s0[r] * QK_SCALE;
            float x1 = s1[r] * QK_SCALE;
            if (skB + l15 > sq)      x0 = NEG_BIG;
            if (skB + 16 + l15 > sq) x1 = NEG_BIG;

            float mx = fmaxf(x0, x1);
#pragma unroll
            for (int off = 8; off >= 1; off >>= 1)
                mx = fmaxf(mx, __shfl_xor(mx, off, 16));
            const float mnew  = fmaxf(mrow[r], mx);
            const float alpha = __expf(mrow[r] - mnew);
            const float e0 = __expf(x0 - mnew);
            const float e1 = __expf(x1 - mnew);
            float rs = e0 + e1;
#pragma unroll
            for (int off = 8; off >= 1; off >>= 1)
                rs += __shfl_xor(rs, off, 16);
            lrow[r] = lrow[r] * alpha + rs;
            mrow[r] = mnew;
            p0[r] = e0; p1[r] = e1;
            o[0][r] *= alpha; o[1][r] *= alpha;
            o[2][r] *= alpha; o[3][r] *= alpha;
        }

        // ---- C-layout -> A-layout via per-wave LDS tile ----
#pragma unroll
        for (int r = 0; r < 8; ++r) {
            const int M = r + 8 * half;
            P[M * 32 + l15]      = f2bf(p0[r]);
            P[M * 32 + 16 + l15] = f2bf(p1[r]);
        }
        asm volatile("s_wait_dscnt 0" ::: "memory");
        const v16bf pa = load_tile_bf16(P + l15 * 32 + kb);

        // ---- PV: P(16x32) x V(32x64) -> 4 N-tiles ----
#pragma unroll
        for (int t = 0; t < 4; ++t)
            o[t] = wmma_bf16(pa, vb[t], o[t]);
    }

    // ---- normalize and scatter to [m = s*B+b][h*64+d] bf16 ----
    const int b = bh >> 4, h = bh & 15;
#pragma unroll
    for (int t = 0; t < 4; ++t) {
#pragma unroll
        for (int r = 0; r < 8; ++r) {
            const int sq = sqB + r + 8 * half;
            const int d  = t * 16 + l15;
            const float val = o[t][r] / lrow[r];
            Xo[(size_t)(sq * BATCH + b) * DM + h * DHD + d] = f2bf(val);
        }
    }
}

// ---------------------------------------------------------------------------
// Stage 3: output projection.  out = Xa @ Wo + bo  (fp32 result)
// Same ping-pong pipelined structure as stage 1, bf16 A operand.
// ---------------------------------------------------------------------------
__global__ __launch_bounds__(32)
void out_proj_kernel(const __bf16* __restrict__ Xa, const __bf16* __restrict__ Wt,
                     const float* __restrict__ bo, float* __restrict__ out)
{
    const int lane  = threadIdx.x;
    const int half  = lane >> 4;
    const int l15   = lane & 15;
    const int kb    = 8 * half;
    const int mBase = blockIdx.x * 16;
    const int nBase = blockIdx.y * 64;

    v8f acc[4] = {};
    const __bf16* arow = Xa + (size_t)(mBase + l15) * DM + kb;
    const __bf16* brow[4];
#pragma unroll
    for (int t = 0; t < 4; ++t)
        brow[t] = Wt + (size_t)(nBase + t * 16 + l15) * DM + kb;

    v16bf a0 = load_tile_bf16(arow);
    v16bf b0[4];
#pragma unroll
    for (int t = 0; t < 4; ++t) b0[t] = load_tile_bf16(brow[t]);
    v16bf a1, b1[4];

    int kk = 0;
    for (; kk < DM - 64; kk += 64) {
        __builtin_prefetch(arow + kk + 96, 0, 3);
        a1 = load_tile_bf16(arow + kk + 32);
#pragma unroll
        for (int t = 0; t < 4; ++t) b1[t] = load_tile_bf16(brow[t] + kk + 32);
#pragma unroll
        for (int t = 0; t < 4; ++t) acc[t] = wmma_bf16(a0, b0[t], acc[t]);
        a0 = load_tile_bf16(arow + kk + 64);
#pragma unroll
        for (int t = 0; t < 4; ++t) b0[t] = load_tile_bf16(brow[t] + kk + 64);
#pragma unroll
        for (int t = 0; t < 4; ++t) acc[t] = wmma_bf16(a1, b1[t], acc[t]);
    }
    a1 = load_tile_bf16(arow + kk + 32);
#pragma unroll
    for (int t = 0; t < 4; ++t) b1[t] = load_tile_bf16(brow[t] + kk + 32);
#pragma unroll
    for (int t = 0; t < 4; ++t) acc[t] = wmma_bf16(a0, b0[t], acc[t]);
#pragma unroll
    for (int t = 0; t < 4; ++t) acc[t] = wmma_bf16(a1, b1[t], acc[t]);

#pragma unroll
    for (int t = 0; t < 4; ++t) {
        const int n = nBase + t * 16 + l15;
        const float bvv = bo[n];
#pragma unroll
        for (int r = 0; r < 8; ++r) {
            const int m = mBase + r + 8 * half;   // row m == flat (s,b) row of output
            out[(size_t)m * DM + n] = acc[t][r] + bvv;
        }
    }
}

// ---------------------------------------------------------------------------
// Launch
// ---------------------------------------------------------------------------
extern "C" void kernel_launch(void* const* d_in, const int* in_sizes, int n_in,
                              void* d_out, int out_size, void* d_ws, size_t ws_size,
                              hipStream_t stream)
{
    const float* query = (const float*)d_in[0];
    const float* key   = (const float*)d_in[1];
    const float* value = (const float*)d_in[2];
    // d_in[3] = causal mask (bool): applied analytically in attn_kernel
    const float* Wq = (const float*)d_in[4];
    const float* bq = (const float*)d_in[5];
    const float* Wk = (const float*)d_in[6];
    const float* bk = (const float*)d_in[7];
    const float* Wv = (const float*)d_in[8];
    const float* bv = (const float*)d_in[9];
    const float* Wo = (const float*)d_in[10];
    const float* bo = (const float*)d_in[11];
    float* out = (float*)d_out;

    const size_t QKV_ELEMS = (size_t)MROWS * DM;   // 4096*1024 bf16 each (8 MB)
    const size_t W_ELEMS   = (size_t)DM * DM;      // 1024*1024 bf16      (2 MB)
    __bf16* qws = (__bf16*)d_ws;
    __bf16* kws = qws + QKV_ELEMS;
    __bf16* vws = kws + QKV_ELEMS;
    __bf16* xws = vws + QKV_ELEMS;
    __bf16* wtq = xws + QKV_ELEMS;
    __bf16* wtk = wtq + W_ELEMS;
    __bf16* wtv = wtk + W_ELEMS;
    __bf16* wto = wtv + W_ELEMS;                   // total 40 MB of ws

    // Stage 0: weights -> bf16, transposed to [n][k]
    dim3 gW(DM / 32, DM / 32, 1);                  // (32, 32)
    wcvt_kernel<<<gW, 256, 0, stream>>>(Wq, wtq);
    wcvt_kernel<<<gW, 256, 0, stream>>>(Wk, wtk);
    wcvt_kernel<<<gW, 256, 0, stream>>>(Wv, wtv);
    wcvt_kernel<<<gW, 256, 0, stream>>>(Wo, wto);

    // Stage 1: QKV projections
    dim3 gGemm(MROWS / 16, DM / 64, 1);            // (256, 16)
    qkv_proj_kernel<<<gGemm, 32, 0, stream>>>(query, wtq, bq, qws, 0);
    qkv_proj_kernel<<<gGemm, 32, 0, stream>>>(key,   wtk, bk, kws, 0);
    qkv_proj_kernel<<<gGemm, 32, 0, stream>>>(value, wtv, bv, vws, 1);

    // Stage 2: 4096 query-tile waves, 4 waves per block -> 1024 blocks
    attn_kernel<<<1024, 128, 0, stream>>>(qws, kws, vws, xws);

    // Stage 3: output projection
    out_proj_kernel<<<gGemm, 32, 0, stream>>>(xws, wto, bo, out);
}
